// FullAttention_DS_integ_woembed_57810259804512
// MI455X (gfx1250) — matrix operand order repaired and checked by reading, not compiled, not run on hardware
//
#include <hip/hip_runtime.h>

#define B_ 8
#define N_ 4096
#define T_ 512
#define M_ 4096

#define QOFF 0ull
#define KOFF 33554432ull     // 32 MB: bf16 Q is B*N*T*2 = 32 MB
#define VOFF 67108864ull     // 64 MB
// total ws use: 96 MB

typedef __attribute__((ext_vector_type(16))) __bf16 v16bf;
typedef __attribute__((ext_vector_type(8)))  float  v8f;
typedef __attribute__((ext_vector_type(4)))  unsigned int v4u;
typedef __attribute__((ext_vector_type(4)))  int    v4i;
typedef __attribute__((ext_vector_type(8)))  int    v8i;

#if defined(__AMDGCN__) && __has_builtin(__builtin_amdgcn_tensor_load_to_lds)
#define HAVE_TDM 1
#else
#define HAVE_TDM 0
#endif

__device__ __forceinline__ unsigned short f2bf(float f) {
  union { float f; unsigned u; } cv; cv.f = f;
  unsigned u = cv.u;
  u += 0x7FFFu + ((u >> 16) & 1u);          // round-to-nearest-even
  return (unsigned short)(u >> 16);
}

// per-lane async 16B copy global -> LDS (ASYNCcnt-tracked)
__device__ __forceinline__ void cp_async16(unsigned lds_off, const void* g) {
  asm volatile("global_load_async_to_lds_b128 %0, %1, off"
               :: "v"(lds_off), "v"((unsigned long long)(size_t)g)
               : "memory");
}
__device__ __forceinline__ void wait_async0() {
  asm volatile("s_wait_asynccnt 0" ::: "memory");
}
__device__ __forceinline__ void wait_tensor0() {
#if defined(__AMDGCN__) && __has_builtin(__builtin_amdgcn_s_wait_tensorcnt)
  __builtin_amdgcn_s_wait_tensorcnt(0);
#else
  asm volatile("s_wait_tensorcnt 0" ::: "memory");
#endif
}

// TDM: one instruction DMAs a linear block of n8 8-byte elements into LDS.
__device__ __forceinline__ void tdm_load_lin(unsigned lds_off, const void* g, unsigned n8) {
#if HAVE_TDM
  const unsigned long long ga = (unsigned long long)(size_t)g;
  // D# group0: count=1 | lds_addr | global_addr[56:0] | type=2
  v4u g0 = { 1u, lds_off, (unsigned)ga,
             (unsigned)((ga >> 32) & 0x01FFFFFFu) | (2u << 30) };
  // D# group1: data_size=3 (8B); tensor_dim0=n8; tensor_dim1=1; tile_dim0=n8; stride0=n8
  v8i g1 = { (int)0x00030000u,
             (int)((n8 & 0xFFFFu) << 16),                    // tensor_dim0 lo16
             (int)(((n8 >> 16) & 0xFFFFu) | (1u << 16)),     // tensor_dim0 hi16 | tensor_dim1=1
             (int)((n8 & 0xFFFFu) << 16),                    // tile_dim0
             0,                                              // tile_dim1/2 unused
             (int)n8, 0, 0 };                                // tensor_dim0_stride
  v4i z4 = {0, 0, 0, 0};
#if __clang_major__ >= 23
  v8i z8 = {0, 0, 0, 0, 0, 0, 0, 0};
  __builtin_amdgcn_tensor_load_to_lds(g0, g1, z4, z4, z8, 0);
#else
  __builtin_amdgcn_tensor_load_to_lds(g0, g1, z4, z4, 0);
#endif
#else
  (void)lds_off; (void)g; (void)n8;
#endif
}

// 16-lane butterfly reductions with immediate-pattern ds_swizzle (no address VALU)
__device__ __forceinline__ float red16_max(float x) {
  x = fmaxf(x, __int_as_float(__builtin_amdgcn_ds_swizzle(__float_as_int(x), (1 << 10) | 31)));
  x = fmaxf(x, __int_as_float(__builtin_amdgcn_ds_swizzle(__float_as_int(x), (2 << 10) | 31)));
  x = fmaxf(x, __int_as_float(__builtin_amdgcn_ds_swizzle(__float_as_int(x), (4 << 10) | 31)));
  x = fmaxf(x, __int_as_float(__builtin_amdgcn_ds_swizzle(__float_as_int(x), (8 << 10) | 31)));
  return x;
}
__device__ __forceinline__ float red16_sum(float x) {
  x += __int_as_float(__builtin_amdgcn_ds_swizzle(__float_as_int(x), (1 << 10) | 31));
  x += __int_as_float(__builtin_amdgcn_ds_swizzle(__float_as_int(x), (2 << 10) | 31));
  x += __int_as_float(__builtin_amdgcn_ds_swizzle(__float_as_int(x), (4 << 10) | 31));
  x += __int_as_float(__builtin_amdgcn_ds_swizzle(__float_as_int(x), (8 << 10) | 31));
  return x;
}

// ---------------- pre-pass: fp32 -> bf16, tiled into WMMA fragment layouts ----------------
// A-frag (16-bit A 16x32): lane = M + 16*K[3]; elem = K[2:0] + 8*K[4]
// B-frag (16-bit B 32x16): lane = N + 16*K[3]; elem = K[2:0] + 8*K[4]

__global__ __launch_bounds__(256) void cvt_q(const float* __restrict__ Qm,
                                             unsigned char* __restrict__ ws) {
  const int i = blockIdx.x * 256 + threadIdx.x;      // B*N*T = 2^24 elements
  const int k = i & (T_ - 1);
  const int n = (i >> 9) & (N_ - 1);
  const int b = i >> 21;
  const float q = Qm[i];
  const int qt16 = n >> 4, r = n & 15;
  const int frag = k >> 5, kk = k & 31;
  const int dl = r + (((kk >> 3) & 1) << 4);
  const int de = (kk & 7) + ((kk >> 4) << 3);
  *(unsigned short*)(ws + QOFF + ((((size_t)b * 256 + qt16) * 16 + frag) << 10)
                     + dl * 32 + de * 2) = f2bf(q);
}

__global__ __launch_bounds__(256) void cvt_k(const float* __restrict__ Km,
                                             unsigned char* __restrict__ ws) {
  const int i = blockIdx.x * 256 + threadIdx.x;      // B*T*M
  const int m = i & (M_ - 1);
  const int t = (i >> 12) & (T_ - 1);
  const int b = i >> 21;
  const float x = Km[i];
  const int mb = m >> 7, ml = m & 127, mt = ml >> 4, nn = ml & 15;
  const int tc = t >> 6, tl = t & 63, kf = tl >> 5, k32 = tl & 31;
  const int dl = nn + (((k32 >> 3) & 1) << 4);
  const int de = (k32 & 7) + ((k32 >> 4) << 3);
  *(unsigned short*)(ws + KOFF + (((((size_t)b * 32 + mb) * 8 + tc) * 16 + mt * 2 + kf) << 10)
                     + dl * 32 + de * 2) = f2bf(x);
}

__global__ __launch_bounds__(256) void cvt_v(const float* __restrict__ Vm,
                                             unsigned char* __restrict__ ws) {
  const int i = blockIdx.x * 256 + threadIdx.x;      // B*T*M
  const int m = i & (M_ - 1);
  const int t = (i >> 12) & (T_ - 1);
  const int b = i >> 21;
  const float x = Vm[i];
  const int mb = m >> 7, kf = (m & 127) >> 5, k32 = m & 31;
  const int tc = t >> 6, tt = (t & 63) >> 4, nn = t & 15;
  const int dl = nn + (((k32 >> 3) & 1) << 4);
  const int de = (k32 & 7) + ((k32 >> 4) << 3);
  *(unsigned short*)(ws + VOFF + (((((size_t)b * 32 + mb) * 8 + tc) * 16 + tt * 4 + kf) << 10)
                     + dl * 32 + de * 2) = f2bf(x);
}

// ---------------- main flash-attention kernel: 32 queries per WG ----------------
__global__ __launch_bounds__(256)
void fa_fwd_wmma_bf16(const unsigned char* __restrict__ Qb,
                      const unsigned char* __restrict__ Kb,
                      const unsigned char* __restrict__ Vb,
                      float* __restrict__ Om) {
  __shared__ __align__(32) unsigned char lds_q[32 * 1024];   // Q 32x512 A-frags
  __shared__ __align__(32) unsigned char lds_kv[16 * 1024];  // K/V chunk B-frags (t=64)
  __shared__ __align__(32) unsigned char lds_p[8 * 1024];    // P 32x128 A-frags
  __shared__ float lds_red[128];
  __shared__ float s_rowm[32], s_rowl[32], s_alpha[32];

  const int tid  = threadIdx.x;
  const int lane = tid & 31;
  const int wv   = tid >> 5;
  const int hi   = lane >> 4;
  const int qr   = wv >> 2;               // query half (16 rows each)
  const int kc   = wv & 3;                // stage-1 key 32-slot == P fragment id
  const int ts   = wv & 3;                // stage-2 t-tile slot
  const int b    = blockIdx.x >> 7;
  const int n0   = (blockIdx.x & 127) << 5;

  const unsigned lq0  = (unsigned)(size_t)&lds_q[0];
  const unsigned lkv0 = (unsigned)(size_t)&lds_kv[0];

  // ---- Q tile: per-lane async copies (32 KB contiguous in pre-tiled layout) ----
  {
    const unsigned char* src = Qb + (((size_t)b * 256 + (n0 >> 4)) << 14);
    #pragma unroll
    for (int u = 0; u < 8; ++u) {
      const int off = u * 4096 + tid * 16;
      cp_async16(lq0 + off, src + off);
    }
  }
  if (tid < 32) { s_rowm[tid] = -1e30f; s_rowl[tid] = 0.0f; }
  wait_async0();
  __syncthreads();

  v8f oacc[8] = {};                       // 8 C-tiles: t = tc*64 + ts*16

  for (int mb = 0; mb < 32; ++mb) {       // 32 key blocks of 128
    v8f sacc0 = {}, sacc1 = {};
    const unsigned char* kbase = Kb + ((((size_t)b * 32 + mb) * 8) << 14);
    const unsigned char* vbase = Vb + ((((size_t)b * 32 + mb) * 8) << 14);

    // ---- stage 1: S(32x128) = Q x K, 8 chunks of t=64 ----
    #pragma unroll 1
    for (int tc = 0; tc < 8; ++tc) {
      __syncthreads();
      const unsigned char* src = kbase + ((size_t)tc << 14);
#if HAVE_TDM
      if (wv == 0) { tdm_load_lin(lkv0, src, 2048); wait_tensor0(); }
#else
      #pragma unroll
      for (int u = 0; u < 4; ++u) {
        const int off = u * 4096 + tid * 16;
        cp_async16(lkv0 + off, src + off);
      }
      wait_async0();
#endif
      __syncthreads();
      #pragma unroll
      for (int kf = 0; kf < 2; ++kf) {
        v16bf a  = *(const v16bf*)(lds_q  + ((qr * 16 + tc * 2 + kf) << 10) + lane * 32);
        v16bf b0 = *(const v16bf*)(lds_kv + ((((kc * 2 + 0) * 2 + kf)) << 10) + lane * 32);
        v16bf b1 = *(const v16bf*)(lds_kv + ((((kc * 2 + 1) * 2 + kf)) << 10) + lane * 32);
        sacc0 = __builtin_amdgcn_wmma_f32_16x16x32_bf16(false, a, false, b0,
                                                        (short)0, sacc0, false, false);
        sacc1 = __builtin_amdgcn_wmma_f32_16x16x32_bf16(false, a, false, b1,
                                                        (short)0, sacc1, false, false);
      }
    }

    // ---- online softmax ----
    float rmax[8];
    #pragma unroll
    for (int v = 0; v < 8; ++v) rmax[v] = red16_max(fmaxf(sacc0[v], sacc1[v]));
    if ((lane & 15) == 0) {
      #pragma unroll
      for (int v = 0; v < 8; ++v) lds_red[(qr * 4 + kc) * 16 + (hi << 3) + v] = rmax[v];
    }
    __syncthreads();
    if (tid < 32) {
      const int q2 = tid >> 4, r = tid & 15;
      float bm = -1e30f;
      #pragma unroll
      for (int u = 0; u < 4; ++u) bm = fmaxf(bm, lds_red[(q2 * 4 + u) * 16 + r]);
      const float mo = s_rowm[tid];
      const float mn = fmaxf(mo, bm);
      s_alpha[tid] = __expf(mo - mn);
      s_rowm[tid]  = mn;
    }
    __syncthreads();

    // rescale O, exponentiate P, row-sum, emit P as bf16 A-fragment (qr, kc)
    float p0[8], p1[8], rsum[8];
    #pragma unroll
    for (int v = 0; v < 8; ++v) {
      const int row = qr * 16 + (hi << 3) + v;
      const float al = s_alpha[row];
      #pragma unroll
      for (int t8 = 0; t8 < 8; ++t8) oacc[t8][v] *= al;
      const float mrow = s_rowm[row];
      p0[v] = __expf(sacc0[v] - mrow);
      p1[v] = __expf(sacc1[v] - mrow);
      rsum[v] = red16_sum(p0[v] + p1[v]);
    }
    {
      const int col = lane & 15;
      #pragma unroll
      for (int v = 0; v < 8; ++v) {
        const int row = (hi << 3) + v;
        {
          const int k32 = col;
          const int dl = row + (((k32 >> 3) & 1) << 4);
          const int de = (k32 & 7) + ((k32 >> 4) << 3);
          *(unsigned short*)(lds_p + ((qr * 4 + kc) << 10) + dl * 32 + de * 2) = f2bf(p0[v]);
        }
        {
          const int k32 = 16 + col;
          const int dl = row + (((k32 >> 3) & 1) << 4);
          const int de = (k32 & 7) + ((k32 >> 4) << 3);
          *(unsigned short*)(lds_p + ((qr * 4 + kc) << 10) + dl * 32 + de * 2) = f2bf(p1[v]);
        }
      }
    }
    if ((lane & 15) == 0) {
      #pragma unroll
      for (int v = 0; v < 8; ++v) lds_red[(qr * 4 + kc) * 16 + (hi << 3) + v] = rsum[v];
    }
    __syncthreads();
    if (tid < 32) {
      const int q2 = tid >> 4, r = tid & 15;
      float bs = 0.0f;
      #pragma unroll
      for (int u = 0; u < 4; ++u) bs += lds_red[(q2 * 4 + u) * 16 + r];
      s_rowl[tid] = s_rowl[tid] * s_alpha[tid] + bs;
    }

    // ---- stage 2: O(32x512) += P(32x128) x V^T; 8 chunks of t=64 ----
    #pragma unroll
    for (int tc = 0; tc < 8; ++tc) {
      __syncthreads();
      const unsigned char* src = vbase + ((size_t)tc << 14);
#if HAVE_TDM
      if (wv == 0) { tdm_load_lin(lkv0, src, 2048); wait_tensor0(); }
#else
      #pragma unroll
      for (int u = 0; u < 4; ++u) {
        const int off = u * 4096 + tid * 16;
        cp_async16(lkv0 + off, src + off);
      }
      wait_async0();
#endif
      __syncthreads();
      #pragma unroll
      for (int kf = 0; kf < 4; ++kf) {
        v16bf a  = *(const v16bf*)(lds_p  + ((qr * 4 + kf) << 10) + lane * 32);
        v16bf bb = *(const v16bf*)(lds_kv + ((ts * 4 + kf) << 10) + lane * 32);
        oacc[tc] = __builtin_amdgcn_wmma_f32_16x16x32_bf16(false, a, false, bb,
                                                           (short)0, oacc[tc], false, false);
      }
    }

    if (mb + 1 < 32) {    // warm L2 for the next key block
      __builtin_prefetch(kbase + (8 << 14) + tid * 64, 0, 1);
      __builtin_prefetch(vbase + (8 << 14) + tid * 64, 0, 1);
    }
  }

  __syncthreads();
  // ---- epilogue: divide by row sums, store O ----
  #pragma unroll
  for (int v = 0; v < 8; ++v) {
    const int row = qr * 16 + (hi << 3) + v;
    const float inv = 1.0f / s_rowl[row];
    const int col = lane & 15;
    const size_t base = ((size_t)b * N_ + n0 + row) * T_;
    #pragma unroll
    for (int tc = 0; tc < 8; ++tc) {
      Om[base + tc * 64 + ts * 16 + col] = oacc[tc][v] * inv;
    }
  }
}

extern "C" void kernel_launch(void* const* d_in, const int* in_sizes, int n_in,
                              void* d_out, int out_size, void* d_ws, size_t ws_size,
                              hipStream_t stream) {
  (void)in_sizes; (void)n_in; (void)out_size; (void)ws_size;
  const float* Q = (const float*)d_in[0];
  const float* K = (const float*)d_in[1];
  const float* V = (const float*)d_in[2];
  float* O = (float*)d_out;
  unsigned char* ws = (unsigned char*)d_ws;

  cvt_q<<<65536, 256, 0, stream>>>(Q, ws);
  cvt_k<<<65536, 256, 0, stream>>>(K, ws);
  cvt_v<<<65536, 256, 0, stream>>>(V, ws);

  fa_fwd_wmma_bf16<<<1024, 256, 0, stream>>>(ws + QOFF, ws + KOFF, ws + VOFF, O);
}
